// ContinuousFilterReadout_90941637526157
// MI455X (gfx1250) — matrix operand order repaired and checked by reading, not compiled, not run on hardware
//
#include <hip/hip_runtime.h>

// ---------------------------------------------------------------------------
// ContinuousFilterReadout fused kernel for gfx1250 (MI455X), wave32 + WMMA
// + Tensor Data Mover (TDM) weight staging into LDS.
//   E = 262144 edges, F = 128, RBF = 32, probes = 100000.
// Block = 256 threads (8 waves); each wave processes a 16-edge tile with
// v_wmma_f32_16x16x32_bf16 (M=16 rows = edges).
// Wave 0 issues one TENSOR_LOAD_TO_LDS moving the 90KB contiguous block of
// later-layer bf16 weights (w_s2T|w_f1T|w_f2T|w_o1T) into LDS, overlapped
// with the layer-1 GEMM; synced via s_wait_tensorcnt before layer 2.
// ---------------------------------------------------------------------------

typedef __attribute__((ext_vector_type(16))) __bf16 v16bf;
typedef __attribute__((ext_vector_type(8)))  __bf16 v8bf;
typedef __attribute__((ext_vector_type(8)))  float  v8f;
typedef __attribute__((ext_vector_type(4)))  unsigned int u32x4;
typedef __attribute__((ext_vector_type(8)))  int i32x8;
typedef __attribute__((ext_vector_type(4)))  int i32x4;

#define FD   128
#define PI_OVER_4 0.7853981633974483f

// later-layer weight block: [ w_s2T (16384) | w_f1T (4096) | w_f2T (16384) | w_o1T (8192) ]
#define WLATER_ELEMS 45056        // bf16 elements  (= 90112 bytes)
#define SH_S2 0
#define SH_F1 16384
#define SH_F2 20480
#define SH_O1 36864

static __device__ __forceinline__ float silu_f(float x) {
    return x / (1.0f + __expf(-x));
}

static __device__ __forceinline__ v8f wmma_bf16(v16bf a, v16bf b, v8f c) {
    return __builtin_amdgcn_wmma_f32_16x16x32_bf16(false, a, false, b,
                                                   (short)0, c, false, false);
}

// build a 16x32 bf16 A-fragment from two contiguous bf16 chunks
static __device__ __forceinline__ v16bf make_a(v8bf lo, v8bf hi) {
    v16bf a;
    #pragma unroll
    for (int i = 0; i < 8; ++i) { a[i] = lo[i]; a[8 + i] = hi[i]; }
    return a;
}

// ---------------------------------------------------------------------------
// Prep kernels
// ---------------------------------------------------------------------------
__global__ void transpose_w_bf16(const float* __restrict__ src,  // [K][N] f32
                                 __bf16* __restrict__ dst,       // [N][K] bf16
                                 int K, int N) {
    int idx = blockIdx.x * blockDim.x + threadIdx.x;
    if (idx < K * N) {
        int n = idx / K;
        int k = idx - n * K;
        dst[idx] = (__bf16)src[k * N + n];
    }
}

// S_bf[a][c] = bf16(S[a][c]);  n_bf[a][c] = bf16(sqrt(sum_d V[a][d][c]^2 + 1e-8))
__global__ void atom_tables_bf16(const float* __restrict__ S,   // [N][128]
                                 const float* __restrict__ V,   // [N][3][128]
                                 __bf16* __restrict__ S_bf,     // [N][128]
                                 __bf16* __restrict__ n_bf,     // [N][128]
                                 int total) {                   // N*128
    int idx = blockIdx.x * blockDim.x + threadIdx.x;
    if (idx < total) {
        int a = idx >> 7;
        int c = idx & 127;
        S_bf[idx] = (__bf16)S[idx];
        const float* Vrow = V + (size_t)a * (3 * FD);
        float v0 = Vrow[c], v1 = Vrow[FD + c], v2 = Vrow[2 * FD + c];
        n_bf[idx] = (__bf16)__builtin_sqrtf(v0 * v0 + v1 * v1 + v2 * v2 + 1e-8f);
    }
}

__global__ void init_out_kernel(float* __restrict__ out,
                                const float* __restrict__ final_bias, int n) {
    int i = blockIdx.x * blockDim.x + threadIdx.x;
    if (i < n) out[i] = final_bias[0];
}

// ---------------------------------------------------------------------------
// Main fused kernel
// ---------------------------------------------------------------------------
__global__ __launch_bounds__(256)
void cfr_main(const float* __restrict__ diff,      // [E][3]
              const __bf16* __restrict__ S_bf,     // [N][128] bf16
              const __bf16* __restrict__ n_bf,     // [N][128] bf16
              const float* __restrict__ V,         // [N][3][128] f32
              const int*   __restrict__ atom_idx,  // [E]
              const int*   __restrict__ probe_idx, // [E]
              const __bf16* __restrict__ WT_s1,    // [128][384] bf16 (transposed)
              const __bf16* __restrict__ Wlater,   // 45056-elem contiguous bf16 block
              const float* __restrict__ b_s1,
              const float* __restrict__ g_ln,
              const float* __restrict__ b_ln,
              const float* __restrict__ b_s2,
              const float* __restrict__ b_f1,
              const float* __restrict__ b_f2,
              const float* __restrict__ b_o1,
              const float* __restrict__ w_o2,      // [64]
              const float* __restrict__ b_o2,      // [1]
              float* __restrict__ out,             // [num_probes]
              int nEdges)
{
    // Per-wave activation scratch for D-layout -> A-layout transposes.
    __shared__ __align__(32) __bf16 shAct[8][16 * FD];        // 32 KB
    // TDM-staged later-layer weights (s2|f1|f2|o1 transposed bf16).
    __shared__ __align__(32) __bf16 shW[WLATER_ELEMS];        // 88 KB

    const int tid  = threadIdx.x;
    const int lane = tid & 31;
    const int wv   = tid >> 5;
    const int half = lane >> 4;      // 0: lanes 0-15, 1: lanes 16-31
    const int mrow = lane & 15;      // row within 16x16 tile / edge-in-tile

    // ---------------- TDM: async-copy later-layer weights to LDS ----------
    // D# descriptor per CDNA5 ISA section 8 (2D tensor, one 45056-elem row,
    // data_size = 2 bytes). Issued once by wave 0; overlaps layer-1 GEMM.
    if (wv == 0) {
        unsigned ldsOff = (unsigned)(uintptr_t)(&shW[0]);
        unsigned long long ga = (unsigned long long)(uintptr_t)Wlater;
        u32x4 g0;
        g0[0] = 1u;                                   // count=1 (valid user D#)
        g0[1] = ldsOff;                               // lds_addr
        g0[2] = (unsigned)(ga & 0xFFFFFFFFu);         // global_addr[31:0]
        g0[3] = (unsigned)((ga >> 32) & 0x01FFFFFFu)  // global_addr[56:32]
              | (2u << 30);                           // type=2 ("image")
        i32x8 g1;
        g1[0] = 0x10000;                              // data_size=1 -> 2 bytes
        g1[1] = (int)((WLATER_ELEMS & 0xFFFF) << 16); // tensor_dim0[15:0]
        g1[2] = (int)((WLATER_ELEMS >> 16) | (1 << 16)); // td0[31:16] | td1[15:0]
        g1[3] = (int)(((unsigned)(WLATER_ELEMS & 0xFFFF)) << 16); // tile_dim0
        g1[4] = 1;                                    // tile_dim1=1, tile_dim2=0
        g1[5] = WLATER_ELEMS;                         // tensor_dim0_stride[31:0]
        g1[6] = 0;
        g1[7] = 0;
        i32x4 g2 = {};
        i32x4 g3 = {};
        i32x8 g4 = {};
        __builtin_amdgcn_tensor_load_to_lds(g0, g1, g2, g3, g4, 0);
    }

    int e = blockIdx.x * 128 + wv * 16 + mrow;
    bool valid = (e < nEdges);
    int eL = valid ? e : (nEdges - 1);

    // ---------------- edge scalars ----------------
    float dx = diff[3 * eL + 0];
    float dy = diff[3 * eL + 1];
    float dz = diff[3 * eL + 2];
    float sq   = dx * dx + dy * dy + dz * dz;
    float dist = __builtin_sqrtf(sq);
    float invs = 1.0f / __builtin_sqrtf(sq + 1e-8f);
    float rx = dx * invs, ry = dy * invs, rz = dz * invs;
    int atom  = atom_idx[eL];
    int probe = probe_idx[eL];

    // polynomial envelope p=5: 1 - 21 x^5 + 35 x^6 - 15 x^7
    float xx = dist * 0.25f;
    float x2 = xx * xx;
    float x5 = x2 * x2 * xx;
    float env = 1.0f - 21.0f * x5 + 35.0f * x5 * xx - 15.0f * x5 * x2;
    env = (dist < 4.0f) ? env : 0.0f;

    // ---------------- RBF A-fragment (K = 32, one k-step) ----------------
    // A 16x32 bf16, this code's contiguous k order (B uses same): lane holds row mrow; K chunks:
    //   half 0 -> K in [0,16); half 1 -> K in [16,32)
    float invd = 1.0f / dist;
    v16bf aRBF;
    #pragma unroll
    for (int i = 0; i < 16; ++i) {
        int K = half * 16 + i;
        aRBF[i] = (__bf16)(__sinf(dist * (float)(K + 1) * PI_OVER_4) * invd);
    }

    const __bf16* Srow = S_bf + (size_t)atom * FD;
    const __bf16* Nrow = n_bf + (size_t)atom * FD;
    const float*  Vrow = V    + (size_t)atom * (3 * FD);

    // ======================= state branch, layer 1 =======================
    // u = [S_e | n_e | q_pi] @ w_s1 + b_s1     (K = 384 -> 12 k-steps)
    // B-fragments from global (L2-resident); overlaps the TDM transfer.
    v8f accS[8];
    #pragma unroll
    for (int n = 0; n < 8; ++n) accS[n] = (v8f){};

    // k-steps 0..3 : S_e columns (direct bf16 gather, 2x16B per k-step)
    for (int j = 0; j < 4; ++j) {
        int c0 = j * 32 + half * 16;
        v16bf a = make_a(*(const v8bf*)(Srow + c0), *(const v8bf*)(Srow + c0 + 8));
        #pragma unroll
        for (int n = 0; n < 8; ++n) {
            const __bf16* bp = WT_s1 + (n * 16 + mrow) * 384 + j * 32 + half * 16;
            accS[n] = wmma_bf16(a, *(const v16bf*)bp, accS[n]);
        }
    }
    // k-steps 4..7 : n_e (precomputed per-atom bf16 table)
    for (int j = 0; j < 4; ++j) {
        int c0 = j * 32 + half * 16;
        v16bf a = make_a(*(const v8bf*)(Nrow + c0), *(const v8bf*)(Nrow + c0 + 8));
        #pragma unroll
        for (int n = 0; n < 8; ++n) {
            const __bf16* bp = WT_s1 + (n * 16 + mrow) * 384 + 128 + j * 32 + half * 16;
            accS[n] = wmma_bf16(a, *(const v16bf*)bp, accS[n]);
        }
    }
    // k-steps 8..11 : q_pi = sum_d V[d] * rhat[d]  (edge-dependent)
    for (int j = 0; j < 4; ++j) {
        int c0 = j * 32 + half * 16;
        v16bf a;
        #pragma unroll
        for (int i = 0; i < 8; ++i) {
            int c = c0 + i;
            a[i] = (__bf16)(Vrow[c] * rx + Vrow[FD + c] * ry + Vrow[2 * FD + c] * rz);
        }
        #pragma unroll
        for (int i = 0; i < 8; ++i) {
            int c = c0 + 8 + i;
            a[8 + i] = (__bf16)(Vrow[c] * rx + Vrow[FD + c] * ry + Vrow[2 * FD + c] * rz);
        }
        #pragma unroll
        for (int n = 0; n < 8; ++n) {
            const __bf16* bp = WT_s1 + (n * 16 + mrow) * 384 + 256 + j * 32 + half * 16;
            accS[n] = wmma_bf16(a, *(const v16bf*)bp, accS[n]);
        }
    }

    // bias
    #pragma unroll
    for (int n = 0; n < 8; ++n) {
        float bb = b_s1[n * 16 + mrow];
        #pragma unroll
        for (int v = 0; v < 8; ++v) accS[n][v] += bb;
    }

    // ---------------- LayerNorm over F=128 per edge row ----------------
    // C/D layout: acc[n][v] = u[row v+8*half][col n*16+mrow]; reduce across
    // the 16 lanes of this half + the 8 n-tiles.
    float mv[8], sv[8];
    #pragma unroll
    for (int v = 0; v < 8; ++v) { mv[v] = 0.0f; sv[v] = 0.0f; }
    #pragma unroll
    for (int n = 0; n < 8; ++n)
        #pragma unroll
        for (int v = 0; v < 8; ++v) { float t = accS[n][v]; mv[v] += t; sv[v] += t * t; }
    #pragma unroll
    for (int off = 1; off < 16; off <<= 1) {
        #pragma unroll
        for (int v = 0; v < 8; ++v) {
            mv[v] += __shfl_xor(mv[v], off, 32);
            sv[v] += __shfl_xor(sv[v], off, 32);
        }
    }
    float mean[8], rstd[8];
    #pragma unroll
    for (int v = 0; v < 8; ++v) {
        mean[v] = mv[v] * (1.0f / 128.0f);
        float var = sv[v] * (1.0f / 128.0f) - mean[v] * mean[v];
        rstd[v] = __builtin_amdgcn_rsqf(var + 1e-5f);
    }

    // normalize + affine + SiLU, write bf16 activation row-major to LDS
    #pragma unroll
    for (int n = 0; n < 8; ++n) {
        int c = n * 16 + mrow;
        float g = g_ln[c], bb = b_ln[c];
        #pragma unroll
        for (int v = 0; v < 8; ++v) {
            float t = (accS[n][v] - mean[v]) * rstd[v] * g + bb;
            shAct[wv][(v + 8 * half) * FD + c] = (__bf16)silu_f(t);
        }
    }

    // TDM transfer must be complete before reading shW; barrier also orders
    // the shAct stores above.
    __builtin_amdgcn_s_wait_tensorcnt(0);
    __syncthreads();

    // ======================= state branch, layer 2 =======================
    v8f accT[8];
    #pragma unroll
    for (int n = 0; n < 8; ++n) accT[n] = (v8f){};
    for (int j = 0; j < 4; ++j) {
        const __bf16* ap = &shAct[wv][mrow * FD + j * 32 + half * 16];
        v16bf a = make_a(*(const v8bf*)ap, *(const v8bf*)(ap + 8));
        #pragma unroll
        for (int n = 0; n < 8; ++n) {
            const __bf16* bp = &shW[SH_S2 + (n * 16 + mrow) * 128 + j * 32 + half * 16];
            accT[n] = wmma_bf16(a, *(const v16bf*)bp, accT[n]);
        }
    }
    #pragma unroll
    for (int n = 0; n < 8; ++n) {
        float bb = b_s2[n * 16 + mrow];
        #pragma unroll
        for (int v = 0; v < 8; ++v) accT[n][v] += bb;   // accT = "state"
    }

    // ======================= filter branch =======================
    v8f accF[8];
    #pragma unroll
    for (int n = 0; n < 8; ++n) accF[n] = (v8f){};
    #pragma unroll
    for (int n = 0; n < 8; ++n) {
        const __bf16* bp = &shW[SH_F1 + (n * 16 + mrow) * 32 + half * 16];
        accF[n] = wmma_bf16(aRBF, *(const v16bf*)bp, accF[n]);
    }
    asm volatile("" ::: "memory");
    #pragma unroll
    for (int n = 0; n < 8; ++n) {
        int c = n * 16 + mrow;
        float bb = b_f1[c];
        #pragma unroll
        for (int v = 0; v < 8; ++v) {
            float t = silu_f(accF[n][v] + bb);
            shAct[wv][(v + 8 * half) * FD + c] = (__bf16)t;
        }
    }
    asm volatile("" ::: "memory");

    v8f accW[8];
    #pragma unroll
    for (int n = 0; n < 8; ++n) accW[n] = (v8f){};
    for (int j = 0; j < 4; ++j) {
        const __bf16* ap = &shAct[wv][mrow * FD + j * 32 + half * 16];
        v16bf a = make_a(*(const v8bf*)ap, *(const v8bf*)(ap + 8));
        #pragma unroll
        for (int n = 0; n < 8; ++n) {
            const __bf16* bp = &shW[SH_F2 + (n * 16 + mrow) * 128 + j * 32 + half * 16];
            accW[n] = wmma_bf16(a, *(const v16bf*)bp, accW[n]);
        }
    }

    // ======================= fuse: P = W * state =======================
    asm volatile("" ::: "memory");
    #pragma unroll
    for (int n = 0; n < 8; ++n) {
        int c = n * 16 + mrow;
        float bb = b_f2[c];
        #pragma unroll
        for (int v = 0; v < 8; ++v) {
            float p = (accW[n][v] + bb) * accT[n][v];
            shAct[wv][(v + 8 * half) * FD + c] = (__bf16)p;
        }
    }
    asm volatile("" ::: "memory");

    // ======================= output head =======================
    v8f accO[4];
    #pragma unroll
    for (int n = 0; n < 4; ++n) accO[n] = (v8f){};
    for (int j = 0; j < 4; ++j) {
        const __bf16* ap = &shAct[wv][mrow * FD + j * 32 + half * 16];
        v16bf a = make_a(*(const v8bf*)ap, *(const v8bf*)(ap + 8));
        #pragma unroll
        for (int n = 0; n < 4; ++n) {
            const __bf16* bp = &shW[SH_O1 + (n * 16 + mrow) * 128 + j * 32 + half * 16];
            accO[n] = wmma_bf16(a, *(const v16bf*)bp, accO[n]);
        }
    }

    // y = silu(accO + b_o1); m_row = y . w_o2 reduced over 64 cols
    float msum[8];
    #pragma unroll
    for (int v = 0; v < 8; ++v) msum[v] = 0.0f;
    #pragma unroll
    for (int n = 0; n < 4; ++n) {
        int c = n * 16 + mrow;
        float bb = b_o1[c];
        float w2 = w_o2[c];
        #pragma unroll
        for (int v = 0; v < 8; ++v) {
            float t = silu_f(accO[n][v] + bb);
            msum[v] += t * w2;
        }
    }
    #pragma unroll
    for (int off = 1; off < 16; off <<= 1) {
        #pragma unroll
        for (int v = 0; v < 8; ++v) msum[v] += __shfl_xor(msum[v], off, 32);
    }

    // one writer lane per edge row: row mrow handled when (mrow>>3)==half
    if (((mrow >> 3) == half) && valid) {
        int v = lane & 7;                 // == mrow & 7
        float m = msum[v] + b_o2[0];
        atomicAdd(&out[probe], m * env);
    }
}

// ---------------------------------------------------------------------------
// Host launcher
// ---------------------------------------------------------------------------
extern "C" void kernel_launch(void* const* d_in, const int* in_sizes, int n_in,
                              void* d_out, int out_size, void* d_ws, size_t ws_size,
                              hipStream_t stream) {
    const float* diff      = (const float*)d_in[0];
    const float* S_JK      = (const float*)d_in[1];
    const float* V_JK      = (const float*)d_in[2];
    const int*   atom_idx  = (const int*)  d_in[3];
    const int*   probe_idx = (const int*)  d_in[4];
    // d_in[5] = num_probes (scalar) -- use out_size instead
    const float* w_s1 = (const float*)d_in[6];
    const float* b_s1 = (const float*)d_in[7];
    const float* g_ln = (const float*)d_in[8];
    const float* b_ln = (const float*)d_in[9];
    const float* w_s2 = (const float*)d_in[10];
    const float* b_s2 = (const float*)d_in[11];
    const float* w_f1 = (const float*)d_in[12];
    const float* b_f1 = (const float*)d_in[13];
    const float* w_f2 = (const float*)d_in[14];
    const float* b_f2 = (const float*)d_in[15];
    const float* w_o1 = (const float*)d_in[16];
    const float* b_o1 = (const float*)d_in[17];
    const float* w_o2 = (const float*)d_in[18];
    const float* b_o2 = (const float*)d_in[19];
    const float* final_bias = (const float*)d_in[20];

    const int E      = in_sizes[0] / 3;
    const int nAtoms = in_sizes[1] / FD;

    // workspace layout (bf16 element offsets, all 32B-aligned)
    // [ WT_s1 | WT_s2 | WT_f1 | WT_f2 | WT_o1 | S_bf | n_bf ]
    // WT_s2..WT_o1 form the contiguous 45056-elem "Wlater" block for the TDM.
    __bf16* ws = (__bf16*)d_ws;
    __bf16* WT_s1 = ws + 0;        // 128*384 = 49152
    __bf16* WT_s2 = ws + 49152;    // 128*128 = 16384   (Wlater + 0)
    __bf16* WT_f1 = ws + 65536;    // 128*32  = 4096    (Wlater + 16384)
    __bf16* WT_f2 = ws + 69632;    // 128*128 = 16384   (Wlater + 20480)
    __bf16* WT_o1 = ws + 86016;    // 64*128  = 8192    (Wlater + 36864)
    __bf16* S_bf  = ws + 94208;    // nAtoms*128
    __bf16* n_bf  = S_bf + (size_t)nAtoms * FD;

    // prep: transpose weights to bf16 (stream-ordered before main kernel)
    transpose_w_bf16<<<(384 * 128 + 255) / 256, 256, 0, stream>>>(w_s1, WT_s1, 384, 128);
    transpose_w_bf16<<<(128 * 128 + 255) / 256, 256, 0, stream>>>(w_s2, WT_s2, 128, 128);
    transpose_w_bf16<<<( 32 * 128 + 255) / 256, 256, 0, stream>>>(w_f1, WT_f1,  32, 128);
    transpose_w_bf16<<<(128 * 128 + 255) / 256, 256, 0, stream>>>(w_f2, WT_f2, 128, 128);
    transpose_w_bf16<<<(128 *  64 + 255) / 256, 256, 0, stream>>>(w_o1, WT_o1, 128,  64);

    // prep: per-atom bf16 tables (S and |V|)
    int atTotal = nAtoms * FD;
    atom_tables_bf16<<<(atTotal + 255) / 256, 256, 0, stream>>>(S_JK, V_JK, S_bf, n_bf, atTotal);

    // out = final_bias (segment_sum base)
    init_out_kernel<<<(out_size + 255) / 256, 256, 0, stream>>>((float*)d_out, final_bias, out_size);

    // main fused kernel: 128 edges per block (8 waves x 16-edge tiles)
    int grid = (E + 127) / 128;
    cfr_main<<<grid, 256, 0, stream>>>(diff, S_bf, n_bf, V_JK, atom_idx, probe_idx,
                                       WT_s1, WT_s2 /*Wlater*/,
                                       b_s1, g_ln, b_ln, b_s2,
                                       b_f1, b_f2, b_o1, w_o2, b_o2,
                                       (float*)d_out, E);
}
